// SFGAT_POI_SVI_16939351015635
// MI455X (gfx1250) — matrix operand order repaired and checked by reading, not compiled, not run on hardware
//
#include <hip/hip_runtime.h>
#include <hip/hip_bf16.h>
#include <stdint.h>

#define NNODES 65536
#define NEDGES 1048576
#define ETOT   (NEDGES + NNODES)

typedef __attribute__((ext_vector_type(16))) __bf16        v16bf;
typedef __attribute__((ext_vector_type(8)))  float         v8f;
typedef __attribute__((ext_vector_type(4)))  unsigned int  v4u;

// ---------------------------------------------------------------------------
// float -> bf16 bits (round to nearest even)
__device__ __host__ inline uint16_t f2bf_bits(float f) {
    union { float f; uint32_t u; } v; v.f = f;
    uint32_t r = v.u + 0x7FFFu + ((v.u >> 16) & 1u);
    return (uint16_t)(r >> 16);
}

// monotone uint key for float atomic-max
__device__ inline unsigned fkey(float f) {
    unsigned u = __float_as_uint(f);
    return (u & 0x80000000u) ? ~u : (u | 0x80000000u);
}
__device__ inline float funkey(unsigned k) {
    unsigned u = (k & 0x80000000u) ? (k & 0x7FFFFFFFu) : ~k;
    return __uint_as_float(u);
}

// ---------------------------------------------------------------------------
// Convert A[:, koff:koff+K] (fp32, row-major, stride lda) into zero-padded
// packed bf16: out[n*KPd + d] = {bf16(A[n,2d]), bf16(A[n,2d+1])}, KPd = KC*16.
__global__ void conv_bf16(const float* __restrict__ A, int lda, int koff, int K,
                          uint32_t* __restrict__ out, int KPd) {
    int idx = blockIdx.x * blockDim.x + threadIdx.x;
    if (idx >= NNODES * KPd) return;
    int n = idx / KPd, d = idx - n * KPd;
    int k0 = 2 * d;
    const float* ar = A + (size_t)n * lda + koff;
    float a = (k0     < K) ? ar[k0]     : 0.0f;
    float b = (k0 + 1 < K) ? ar[k0 + 1] : 0.0f;
    out[idx] = ((uint32_t)f2bf_bits(b) << 16) | f2bf_bits(a);
}

// ---------------------------------------------------------------------------
// Pack weights W[K,F] (row-major fp32) into per-lane B-fragment order for
// v_wmma_f32_16x16x32_bf16.  B tile (nt,kc) is 32x16:
//   lane<16  : col = nt*16+lane,    K rows kc*32 + 0..7, 16..23  (2 per dword)
//   lane>=16 : col = nt*16+lane-16, K rows kc*32 + 8..15, 24..31
// index = ((nt*KC + kc)*32 + lane)*8 + j ; dword j = {K=kb+2(j&3)+16(j>>2), +1}
__global__ void pack_w(const float* __restrict__ W, uint32_t* __restrict__ P,
                       int K, int F, int KC) {
    int idx = blockIdx.x * blockDim.x + threadIdx.x;
    int total = (F / 16) * KC * 256;
    if (idx >= total) return;
    int j    = idx & 7;
    int lane = (idx >> 3) & 31;
    int rest = idx >> 8;
    int kc   = rest % KC;
    int nt   = rest / KC;
    int n    = nt * 16 + (lane & 15);
    int kb   = kc * 32 + ((lane >> 4) << 3);
    int k0 = kb + 2 * (j & 3) + ((j >> 2) << 4), k1 = k0 + 1;
    uint32_t lo = (k0 < K) ? f2bf_bits(W[(size_t)k0 * F + n]) : 0;
    uint32_t hi = (k1 < K) ? f2bf_bits(W[(size_t)k1 * F + n]) : 0;
    P[idx] = (hi << 16) | lo;
}

// ---------------------------------------------------------------------------
// C[rows, coff:coff+F] = act(Abf @ W + bias).   One wave computes a 16x64
// output strip (4 column tiles): A fragment loaded once per K-chunk, 4
// independent WMMAs.  KC is a template param so the K loop fully unrolls.
template <int KC>
__global__ __launch_bounds__(256)
void gemm_wmma_bf16(const uint32_t* __restrict__ Abf, int KPd,
                    const uint32_t* __restrict__ Wp,
                    const float* __restrict__ bias,
                    float* __restrict__ Cout, int ldo, int coff,
                    int Fgroups, int relu) {
    int wave = (int)((blockIdx.x * blockDim.x + threadIdx.x) >> 5);
    int lane = threadIdx.x & 31;
    int groups = (NNODES / 16) * Fgroups;
    if (wave >= groups) return;
    int mt = wave / Fgroups;
    int ng = wave - mt * Fgroups;

    // per-lane A base: row = mt*16 + (lane&15); dword offset (lane>=16 ? 4 : 0)
    const uint32_t* ar0 = Abf + (size_t)(mt * 16 + (lane & 15)) * KPd
                              + ((lane >> 4) << 2);
    const uint32_t* wp0 = Wp + ((size_t)(ng * 4) * KC + 0) * 256 + lane * 8;

    v8f acc[4] = {v8f{}, v8f{}, v8f{}, v8f{}};
#pragma unroll
    for (int kc = 0; kc < KC; ++kc) {
        union { v4u q[2]; v16bf bf; } ua;
        const uint32_t* ar = ar0 + kc * 16;
        ua.q[0] = *(const v4u*)ar;          // K = kb0 + 0..7
        ua.q[1] = *(const v4u*)(ar + 8);    // K = kb0 + 16..23
        v16bf af = ua.bf;
#pragma unroll
        for (int t = 0; t < 4; ++t) {
            union { v4u q[2]; v16bf bf; } ub;
            const uint32_t* wp = wp0 + ((size_t)t * KC + kc) * 256;
            ub.q[0] = *(const v4u*)wp;
            ub.q[1] = *(const v4u*)(wp + 4);
            acc[t] = __builtin_amdgcn_wmma_f32_16x16x32_bf16(
                         false, af, false, ub.bf, (short)0, acc[t], false, false);
        }
    }

    int rbase = (lane >> 4) << 3;           // C layout: lanes>=16 -> rows +8
#pragma unroll
    for (int t = 0; t < 4; ++t) {
        int colw = (ng * 4 + t) * 16 + (lane & 15);
        float bv = bias ? bias[colw] : 0.0f;
#pragma unroll
        for (int r = 0; r < 8; ++r) {
            float v = acc[t][r] + bv;
            if (relu) v = fmaxf(v, 0.0f);
            Cout[(size_t)(mt * 16 + rbase + r) * ldo + coff + colw] = v;
        }
    }
}

// ---------------------------------------------------------------------------
__global__ void zero_u32(uint32_t* p, size_t n) {
    size_t i = (size_t)blockIdx.x * blockDim.x + threadIdx.x;
    if (i < n) p[i] = 0;
}

// s[n] = h[n]·asrc , t[n] = h[n]·adst  (wave per node)
__global__ void gat_st(const float* __restrict__ hW, const float* __restrict__ as_,
                       const float* __restrict__ ad_, float* __restrict__ s,
                       float* __restrict__ t, int F) {
    int wave = (int)((blockIdx.x * blockDim.x + threadIdx.x) >> 5);
    int lane = threadIdx.x & 31;
    if (wave >= NNODES) return;
    const float* hr = hW + (size_t)wave * F;
    float ss = 0.f, tt = 0.f;
    for (int j = lane; j < F; j += 32) {
        float hv = hr[j];
        ss += hv * as_[j];
        tt += hv * ad_[j];
    }
#pragma unroll
    for (int off = 16; off; off >>= 1) {
        ss += __shfl_down(ss, off, 32);
        tt += __shfl_down(tt, off, 32);
    }
    if (lane == 0) { s[wave] = ss; t[wave] = tt; }
}

__global__ void edge_pass1(const int* __restrict__ ei, const float* __restrict__ s,
                           const float* __restrict__ t, float* __restrict__ ev,
                           unsigned* __restrict__ mkey) {
    int e = blockIdx.x * blockDim.x + threadIdx.x;
    if (e >= ETOT) return;
    int sn, dn;
    if (e < NEDGES) { sn = ei[e]; dn = ei[NEDGES + e]; } else { sn = dn = e - NEDGES; }
    float v = s[sn] + t[dn];
    v = (v > 0.0f) ? v : 0.2f * v;          // leaky_relu(0.2)
    ev[e] = v;
    atomicMax(mkey + dn, fkey(v));
}

__global__ void edge_pass2(const int* __restrict__ ei, float* __restrict__ ev,
                           const unsigned* __restrict__ mkey, float* __restrict__ den) {
    int e = blockIdx.x * blockDim.x + threadIdx.x;
    if (e >= ETOT) return;
    int dn = (e < NEDGES) ? ei[NEDGES + e] : e - NEDGES;
    float ee = __expf(ev[e] - funkey(mkey[dn]));
    ev[e] = ee;
    atomicAdd(den + dn, ee);
}

// wave per edge: acc[dst] += h[src] * alpha
__global__ void edge_aggregate(const int* __restrict__ ei, const float* __restrict__ hW,
                               const float* __restrict__ ev, const float* __restrict__ den,
                               float* __restrict__ acc, int F) {
    int wave = (int)((blockIdx.x * blockDim.x + threadIdx.x) >> 5);
    int lane = threadIdx.x & 31;
    if (wave >= ETOT) return;
    int sn, dn;
    if (wave < NEDGES) { sn = ei[wave]; dn = ei[NEDGES + wave]; } else { sn = dn = wave - NEDGES; }
    float alpha = ev[wave] / den[dn];
    const float* hs = hW + (size_t)sn * F;
    float* ad = acc + (size_t)dn * F;
    for (int j = lane; j < F; j += 32)
        atomicAdd(ad + j, hs[j] * alpha);
}

__global__ void gat_finalize(const float* __restrict__ acc, const float* __restrict__ b,
                             float* __restrict__ out, int F, int ldo, int coff) {
    int idx = blockIdx.x * blockDim.x + threadIdx.x;
    if (idx >= NNODES * F) return;
    int n = idx / F, j = idx - n * F;
    out[(size_t)n * ldo + coff + j] = fmaxf(acc[idx] + b[j], 0.0f);
}

// gates (N,256): h = sigmoid(o)*tanh(sigmoid(i)*tanh(g))
__global__ void lstm_act(const float* __restrict__ g, float* __restrict__ h) {
    int idx = blockIdx.x * blockDim.x + threadIdx.x;
    if (idx >= NNODES * 64) return;
    int n = idx >> 6, j = idx & 63;
    const float* gr = g + (size_t)n * 256;
    float si = 1.f / (1.f + __expf(-gr[j]));
    float c  = si * tanhf(gr[128 + j]);
    float so = 1.f / (1.f + __expf(-gr[192 + j]));
    h[idx] = so * tanhf(c);
}

// out[n] = z[n,0:64]·w + b  (wave per node)
__global__ void head_dot(const float* __restrict__ z, const float* __restrict__ w,
                         const float* __restrict__ b, float* __restrict__ out) {
    int wave = (int)((blockIdx.x * blockDim.x + threadIdx.x) >> 5);
    int lane = threadIdx.x & 31;
    if (wave >= NNODES) return;
    const float* zr = z + (size_t)wave * 64;
    float v = zr[lane] * w[lane] + zr[lane + 32] * w[lane + 32];
#pragma unroll
    for (int off = 16; off; off >>= 1) v += __shfl_down(v, off, 32);
    if (lane == 0) out[wave] = v + b[0];
}

// ---------------------------------------------------------------------------
extern "C" void kernel_launch(void* const* d_in, const int* in_sizes, int n_in,
                              void* d_out, int out_size, void* d_ws, size_t ws_size,
                              hipStream_t stream) {
    const float* x   = (const float*)d_in[0];
    const int*   ei  = (const int*)d_in[1];
    const float* poi1_W = (const float*)d_in[2];  const float* poi1_b = (const float*)d_in[3];
    const float* poi2_W = (const float*)d_in[4];  const float* poi2_b = (const float*)d_in[5];
    const float* svi1_W = (const float*)d_in[6];  const float* svi1_b = (const float*)d_in[7];
    const float* svi2_W = (const float*)d_in[8];  const float* svi2_b = (const float*)d_in[9];
    const float* all1_W = (const float*)d_in[10]; const float* all1_b = (const float*)d_in[11];
    const float* all2_W = (const float*)d_in[12]; const float* all2_b = (const float*)d_in[13];
    const float* g1W = (const float*)d_in[14]; const float* g1b = (const float*)d_in[15];
    const float* g1as = (const float*)d_in[16]; const float* g1ad = (const float*)d_in[17];
    const float* g2W = (const float*)d_in[18]; const float* g2b = (const float*)d_in[19];
    const float* g2as = (const float*)d_in[20]; const float* g2ad = (const float*)d_in[21];
    const float* g3W = (const float*)d_in[22]; const float* g3b = (const float*)d_in[23];
    const float* g3as = (const float*)d_in[24]; const float* g3ad = (const float*)d_in[25];
    const float* Wih0 = (const float*)d_in[26];
    const float* Wih  = (const float*)d_in[27];
    /* d_in[28] = lstm_Whh : unused by the reference math */
    const float* lstm_b = (const float*)d_in[29];
    const float* t1W = (const float*)d_in[30]; const float* t1b = (const float*)d_in[31];
    const float* t2W = (const float*)d_in[32]; const float* t2b = (const float*)d_in[33];
    const float* l1W = (const float*)d_in[34]; const float* l1b = (const float*)d_in[35];
    const float* l2W = (const float*)d_in[36]; const float* l2b = (const float*)d_in[37];
    const float* l3W = (const float*)d_in[38]; const float* l3b = (const float*)d_in[39];

    // workspace arena (units: 4-byte words)
    float* ws = (float*)d_ws;
    size_t o = 0;
    float* R0 = ws + o; o += (size_t)NNODES * 256;   // hcat (ld192) then lstm gates (ld256)
    float* X1 = ws + o; o += (size_t)NNODES * 128;
    float* X2 = ws + o; o += (size_t)NNODES * 128;
    float* X3 = ws + o; o += (size_t)NNODES * 128;   // aggregation accumulator
    float* Y1 = ws + o; o += (size_t)NNODES * 64;
    float* Y2 = ws + o; o += (size_t)NNODES * 64;
    float* EV = ws + o; o += (size_t)ETOT;
    float* Sv = ws + o; o += NNODES;
    float* Tv = ws + o; o += NNODES;
    float* Dn = ws + o; o += NNODES;
    unsigned* MK = (unsigned*)(ws + o); o += NNODES;
    uint32_t* WP  = (uint32_t*)(ws + o); o += 32768;               // packed weights (128 KB)
    uint32_t* ABF = (uint32_t*)(ws + o); o += (size_t)NNODES * 192; // bf16 A (KP<=384)

    auto pack = [&](const float* W, int K, int F) {
        int KC = (K + 31) / 32;
        int total = (F / 16) * KC * 256;
        pack_w<<<(total + 255) / 256, 256, 0, stream>>>(W, WP, K, F, KC);
    };
    auto gemm = [&](const float* A, int lda, int koff, int K, int F,
                    const float* bias, float* C, int ldo, int coff, int relu) {
        int KC = (K + 31) / 32, KPd = KC * 16;
        size_t tot = (size_t)NNODES * KPd;
        conv_bf16<<<(unsigned)((tot + 255) / 256), 256, 0, stream>>>(A, lda, koff, K, ABF, KPd);
        int Fg = F / 64;
        int groups = (NNODES / 16) * Fg;
        dim3 gr((groups + 7) / 8);
        switch (KC) {
        case 1:  gemm_wmma_bf16<1> <<<gr, 256, 0, stream>>>(ABF, KPd, WP, bias, C, ldo, coff, Fg, relu); break;
        case 2:  gemm_wmma_bf16<2> <<<gr, 256, 0, stream>>>(ABF, KPd, WP, bias, C, ldo, coff, Fg, relu); break;
        case 4:  gemm_wmma_bf16<4> <<<gr, 256, 0, stream>>>(ABF, KPd, WP, bias, C, ldo, coff, Fg, relu); break;
        case 6:  gemm_wmma_bf16<6> <<<gr, 256, 0, stream>>>(ABF, KPd, WP, bias, C, ldo, coff, Fg, relu); break;
        default: gemm_wmma_bf16<12><<<gr, 256, 0, stream>>>(ABF, KPd, WP, bias, C, ldo, coff, Fg, relu); break;
        }
    };
    auto zero = [&](void* p, size_t n) {
        zero_u32<<<(unsigned)((n + 255) / 256), 256, 0, stream>>>((uint32_t*)p, n);
    };
    auto gat = [&](float* hin, const float* W, const float* b, const float* as_,
                   const float* ad_, int Fin, int Fout, float* hW, float* accb,
                   float* outb, int ldo, int coff) {
        pack(W, Fin, Fout);
        gemm(hin, Fin, 0, Fin, Fout, nullptr, hW, Fout, 0, 0);
        gat_st<<<NNODES / 8, 256, 0, stream>>>(hW, as_, ad_, Sv, Tv, Fout);
        zero(MK, NNODES); zero(Dn, NNODES); zero(accb, (size_t)NNODES * Fout);
        edge_pass1<<<ETOT / 256, 256, 0, stream>>>(ei, Sv, Tv, EV, MK);
        edge_pass2<<<ETOT / 256, 256, 0, stream>>>(ei, EV, MK, Dn);
        edge_aggregate<<<(ETOT + 7) / 8, 256, 0, stream>>>(ei, hW, EV, Dn, accb, Fout);
        gat_finalize<<<(NNODES * Fout + 255) / 256, 256, 0, stream>>>(accb, b, outb, Fout, ldo, coff);
    };

    // node-feature MLPs  (x[:,3:16], x[:,56:421] -> hcat[N,192] -> h0[N,128])
    pack(poi1_W, 13, 64);   gemm(x, 445, 3, 13, 64, poi1_b, Y1, 64, 0, 1);
    pack(poi2_W, 64, 64);   gemm(Y1, 64, 0, 64, 64, poi2_b, R0, 192, 0, 1);
    pack(svi1_W, 365, 128); gemm(x, 445, 56, 365, 128, svi1_b, X1, 128, 0, 1);
    pack(svi2_W, 128, 128); gemm(X1, 128, 0, 128, 128, svi2_b, R0, 192, 64, 1);
    pack(all1_W, 192, 128); gemm(R0, 192, 0, 192, 128, all1_b, X1, 128, 0, 1);
    pack(all2_W, 128, 128); gemm(X1, 128, 0, 128, 128, all2_b, X2, 128, 0, 1);

    // GAT stack (h lives in X2; hW scratch X1/Y1; acc X3)
    gat(X2, g1W, g1b, g1as, g1ad, 128, 128, X1, X3, X2, 128, 0);
    gat(X2, g2W, g2b, g2as, g2ad, 128, 128, X1, X3, X2, 128, 0);
    gat(X2, g3W, g3b, g3as, g3ad, 128, 64, Y1, X3, X1, 128, 0);  // z[:,0:64] = X1

    // LSTM: only layers 0..2 are live (t = hs[2]); Whh is dead in the reference
    pack(Wih0, 24, 256);             gemm(x, 445, 421, 24, 256, lstm_b, R0, 256, 0, 0);
    lstm_act<<<NNODES * 64 / 256, 256, 0, stream>>>(R0, Y2);
    pack(Wih, 64, 256);              gemm(Y2, 64, 0, 64, 256, lstm_b + 256, R0, 256, 0, 0);
    lstm_act<<<NNODES * 64 / 256, 256, 0, stream>>>(R0, Y1);
    pack(Wih + (size_t)64 * 256, 64, 256);
                                     gemm(Y1, 64, 0, 64, 256, lstm_b + 512, R0, 256, 0, 0);
    lstm_act<<<NNODES * 64 / 256, 256, 0, stream>>>(R0, Y2);    // t = Y2

    // time MLP -> z[:,64:128]
    pack(t1W, 64, 64); gemm(Y2, 64, 0, 64, 64, t1b, Y1, 64, 0, 1);
    pack(t2W, 64, 64); gemm(Y1, 64, 0, 64, 64, t2b, X1, 128, 64, 1);

    // head
    pack(l1W, 128, 64); gemm(X1, 128, 0, 128, 64, l1b, Y1, 64, 0, 1);
    pack(l2W, 64, 64);  gemm(Y1, 64, 0, 64, 64, l2b, Y2, 64, 0, 1);
    head_dot<<<NNODES / 8, 256, 0, stream>>>(Y2, l3W, l3b, (float*)d_out);
}
